// TXCDRRankKFeature_90984587198482
// MI455X (gfx1250) — compile-verified, matrix operand order checked by reading, not run on hardware
//
#include <hip/hip_runtime.h>
#include <hip/hip_bf16.h>

// ---------------------------------------------------------------------------
// Problem constants (fixed by the reference)
// ---------------------------------------------------------------------------
#define B_SZ   4096
#define T_POS  5
#define D_IN   768
#define D_SAE  16384
#define K_RANK 4
#define K_SEL  64

#define MDIM   B_SZ            // 4096
#define KDIM   (T_POS * D_IN)  // 3840
#define NDIM   D_SAE           // 16384

// GEMM tiling: block 64(M) x 256(N) x 32(K); 8 waves (2x4), wave tile 32x64
#define BM 64
#define BN 256
#define BK 32

typedef __attribute__((ext_vector_type(16))) __bf16 v16bf;
typedef __attribute__((ext_vector_type(8)))  float  v8f;

union Frag {            // one WMMA 16-bit A/B operand: 16 bf16 = 32 bytes / lane
    uint4 q[2];
    v16bf v;
};

// ---------------------------------------------------------------------------
// Kernel 1: pre = X @ W_enc + b_enc   (bf16 WMMA, f32 accumulate)
// grid.x = M-blocks (fast) so concurrent blocks share one 3.9 MB W strip in L2;
// X (63 MB) is L2-resident. Output written into the z-region of d_out.
// ---------------------------------------------------------------------------
__global__ __launch_bounds__(256)
void sae_enc_gemm(const float* __restrict__ X, const float* __restrict__ W,
                  const float* __restrict__ benc, float* __restrict__ pre) {
    // padded rows: 40 bf16 = 80 B -> every 8-elem (16 B) chunk stays 16 B aligned
    __shared__ __align__(16) __bf16 lX[BM][40];   // [m][k]
    __shared__ __align__(16) __bf16 lW[BN][40];   // [n][k]  (transposed)

    const int tid  = threadIdx.x;
    const int lane = tid & 31;
    const int wv   = tid >> 5;        // wave 0..7
    const int wm   = wv >> 2;         // 0..1  (32-row slab)
    const int wn   = wv & 3;          // 0..3  (64-col slab)
    const int hl   = lane >> 4;       // half-wave (K-interleave select)
    const int lr   = lane & 15;       // row (A) / col (B) within 16
    const int mBase = blockIdx.x * BM;
    const int nBase = blockIdx.y * BN;

    v8f acc[2][4] = {};               // eight 16x16 f32 accumulators (32x64)

    for (int k0 = 0; k0 < KDIM; k0 += BK) {
        // ---- stage X tile (64x32 f32 -> bf16), coalesced along K ----
        #pragma unroll
        for (int rep = 0; rep < 2; ++rep) {
            int id = tid + rep * 256;             // 512 float4 total
            int r  = id >> 3, c4 = id & 7;
            const float4 v = *reinterpret_cast<const float4*>(
                X + (size_t)(mBase + r) * KDIM + k0 + c4 * 4);
            __bf16* px = &lX[r][c4 * 4];          // contiguous -> packed ds store
            px[0] = (__bf16)v.x; px[1] = (__bf16)v.y;
            px[2] = (__bf16)v.z; px[3] = (__bf16)v.w;
        }
        // ---- stage W tile (32x256 f32 -> bf16), transposed into [n][k] ----
        #pragma unroll
        for (int rep = 0; rep < 8; ++rep) {
            int id = tid + rep * 256;             // 2048 float4 total
            int kr = id >> 6, c4 = id & 63;
            const float4 v = *reinterpret_cast<const float4*>(
                W + (size_t)(k0 + kr) * NDIM + nBase + c4 * 4);
            lW[c4 * 4 + 0][kr] = (__bf16)v.x;
            lW[c4 * 4 + 1][kr] = (__bf16)v.y;
            lW[c4 * 4 + 2][kr] = (__bf16)v.z;
            lW[c4 * 4 + 3][kr] = (__bf16)v.w;
        }
        // prefetch next K tiles into L2 while we compute (global_prefetch_b8)
        if (k0 + BK < KDIM) {
            __builtin_prefetch(X + (size_t)(mBase + (tid >> 3)) * KDIM + (k0 + BK), 0, 1);
            __builtin_prefetch(W + (size_t)(k0 + BK + (tid >> 5)) * NDIM + nBase + (tid & 31) * 8, 0, 1);
        }
        __syncthreads();

        // ---- build fragments (two ds_load_b128 each) and issue 8 WMMAs ----
        Frag af[2], bf[4];
        #pragma unroll
        for (int mt = 0; mt < 2; ++mt) {
            int r = wm * 32 + mt * 16 + lr;
            af[mt].q[0] = *reinterpret_cast<const uint4*>(&lX[r][hl * 8]);
            af[mt].q[1] = *reinterpret_cast<const uint4*>(&lX[r][16 + hl * 8]);
        }
        #pragma unroll
        for (int nt = 0; nt < 4; ++nt) {
            int cc = wn * 64 + nt * 16 + lr;
            bf[nt].q[0] = *reinterpret_cast<const uint4*>(&lW[cc][hl * 8]);
            bf[nt].q[1] = *reinterpret_cast<const uint4*>(&lW[cc][16 + hl * 8]);
        }
        #pragma unroll
        for (int mt = 0; mt < 2; ++mt)
            #pragma unroll
            for (int nt = 0; nt < 4; ++nt)
                acc[mt][nt] = __builtin_amdgcn_wmma_f32_16x16x32_bf16(
                    false, af[mt].v, false, bf[nt].v,
                    (short)0, acc[mt][nt], false, false);
        __syncthreads();
    }

    // ---- epilogue: add b_enc, store pre ----
    #pragma unroll
    for (int mt = 0; mt < 2; ++mt)
        #pragma unroll
        for (int nt = 0; nt < 4; ++nt) {
            int col = nBase + wn * 64 + nt * 16 + lr;
            float be = benc[col];
            #pragma unroll
            for (int r = 0; r < 8; ++r) {
                int row = mBase + wm * 32 + mt * 16 + hl * 8 + r;
                pre[(size_t)row * NDIM + col] = acc[mt][nt][r] + be;
            }
        }
}

// ---------------------------------------------------------------------------
// Kernel 2: per-row top-64. The row's raw bits are staged LDS-resident ONCE
// via global_load_async_to_lds_b128 (ASYNCcnt), then 4-pass 8-bit radix
// select runs entirely out of LDS; selection rewrites the row in place as
// ReLU'd sparse z and emits a compact (idx,val) list to workspace.
// Dynamic LDS: 16384 keys (64 KB) + 256 histogram bins.
// ---------------------------------------------------------------------------
#define TOPK_LDS_BYTES ((D_SAE + 256) * 4)

__global__ __launch_bounds__(256)
void sae_topk(float* __restrict__ z, int* __restrict__ idxL, float* __restrict__ valL) {
    extern __shared__ unsigned int smem[];
    unsigned int* sBits = smem;          // raw float bits of the row [16384]
    unsigned int* hist  = smem + D_SAE;  // [256]
    __shared__ unsigned int sDigit, sWant, sTie, sCnt;

    const int b   = blockIdx.x;
    const int tid = threadIdx.x;
    float* row = z + (size_t)b * D_SAE;

    // ---- async copy: global row -> LDS (16 x b128 per thread-slice) ----
    #pragma unroll
    for (int it = 0; it < D_SAE / 4 / 256; ++it) {     // 16 issues / thread
        int idx4 = tid + it * 256;
        unsigned int ldsAddr = (unsigned int)(uintptr_t)(sBits + idx4 * 4);
        unsigned long long gAddr = (unsigned long long)(uintptr_t)(row + idx4 * 4);
        asm volatile("global_load_async_to_lds_b128 %0, %1, off"
                     :: "v"(ldsAddr), "v"(gAddr) : "memory");
    }
    asm volatile("s_wait_asynccnt 0" ::: "memory");
    __syncthreads();

    unsigned int prefix = 0, mask = 0, want = K_SEL;

    for (int shift = 24; shift >= 0; shift -= 8) {
        hist[tid] = 0;
        __syncthreads();
        for (int i = tid; i < D_SAE; i += 256) {
            unsigned int u = sBits[i];
            u ^= (u >> 31) ? 0xFFFFFFFFu : 0x80000000u;   // order-preserving map
            if ((u & mask) == prefix)
                atomicAdd(&hist[(u >> shift) & 255u], 1u);
        }
        __syncthreads();
        if (tid == 0) {
            unsigned int cum = 0, dsel = 0, w = want;
            for (int d = 255; d >= 0; --d) {
                unsigned int c = hist[d];
                if (cum + c >= w) { dsel = (unsigned)d; w -= cum; break; }
                cum += c;
            }
            sDigit = dsel; sWant = w;
        }
        __syncthreads();
        prefix |= sDigit << shift;
        want    = sWant;
        mask   |= (0xFFu << shift);
        __syncthreads();
    }

    const unsigned int thr = prefix;   // exact 32-bit key of the k-th largest
    if (tid == 0) { sTie = 0; sCnt = 0; }
    __syncthreads();

    for (int i = tid; i < D_SAE; i += 256) {
        unsigned int bits = sBits[i];
        unsigned int u = bits ^ ((bits >> 31) ? 0xFFFFFFFFu : 0x80000000u);
        bool sel = (u > thr);
        if (!sel && u == thr) {
            unsigned p = atomicAdd(&sTie, 1u);   // cap ties to exactly k
            sel = (p < want);
        }
        float zv = 0.0f;
        if (sel) {
            float f = __builtin_bit_cast(float, bits);
            zv = f > 0.0f ? f : 0.0f;            // ReLU on selected values
            unsigned slot = atomicAdd(&sCnt, 1u);
            idxL[b * K_SEL + slot] = i;
            valL[b * K_SEL + slot] = zv;
        }
        row[i] = zv;                              // pre -> z, in place
    }
}

// ---------------------------------------------------------------------------
// Kernel 3: sparse decode. x_hat[b] = b_dec + sum_f z_bj * (A_j @ B_j).
// Never materializes W_dec (252 MB). One block per batch row; 256 threads,
// each owning 3 columns of d for all 5 positions (15 f32 accumulators).
// ---------------------------------------------------------------------------
__global__ __launch_bounds__(256)
void sae_decode(const float* __restrict__ x, const float* __restrict__ A,
                const float* __restrict__ Bm, const float* __restrict__ bdec,
                const int* __restrict__ idxL, const float* __restrict__ valL,
                float* __restrict__ xhat, float* __restrict__ rowloss) {
    const int b   = blockIdx.x;
    const int tid = threadIdx.x;

    __shared__ float coef[T_POS * K_RANK];   // z_bj * A[j,t,k]
    __shared__ float red[256];

    float acc[T_POS][3];
    #pragma unroll
    for (int t = 0; t < T_POS; ++t)
        #pragma unroll
        for (int di = 0; di < 3; ++di)
            acc[t][di] = bdec[t * D_IN + tid + di * 256];

    for (int f = 0; f < K_SEL; ++f) {
        __syncthreads();                          // protect coef from prev iter
        int j = idxL[b * K_SEL + f];              // uniform -> broadcast load
        if (tid < T_POS * K_RANK)
            coef[tid] = valL[b * K_SEL + f] * A[(size_t)j * (T_POS * K_RANK) + tid];
        __syncthreads();

        const float* Bj = Bm + (size_t)j * (K_RANK * D_IN);
        #pragma unroll
        for (int kk = 0; kk < K_RANK; ++kk) {
            float c0 = coef[0 * K_RANK + kk];
            float c1 = coef[1 * K_RANK + kk];
            float c2 = coef[2 * K_RANK + kk];
            float c3 = coef[3 * K_RANK + kk];
            float c4 = coef[4 * K_RANK + kk];
            #pragma unroll
            for (int di = 0; di < 3; ++di) {
                float bv = Bj[kk * D_IN + tid + di * 256];   // mostly L2 hits
                acc[0][di] = fmaf(c0, bv, acc[0][di]);
                acc[1][di] = fmaf(c1, bv, acc[1][di]);
                acc[2][di] = fmaf(c2, bv, acc[2][di]);
                acc[3][di] = fmaf(c3, bv, acc[3][di]);
                acc[4][di] = fmaf(c4, bv, acc[4][di]);
            }
        }
    }

    float lsum = 0.0f;
    #pragma unroll
    for (int t = 0; t < T_POS; ++t)
        #pragma unroll
        for (int di = 0; di < 3; ++di) {
            int d = tid + di * 256;
            size_t o = (size_t)b * (T_POS * D_IN) + t * D_IN + d;
            float xh = acc[t][di];
            xhat[o] = xh;
            float df = xh - x[o];
            lsum = fmaf(df, df, lsum);
        }

    red[tid] = lsum;
    __syncthreads();
    for (int s = 128; s > 0; s >>= 1) {
        if (tid < s) red[tid] += red[tid + s];
        __syncthreads();
    }
    if (tid == 0) rowloss[b] = red[0];
}

// ---------------------------------------------------------------------------
// Kernel 4: loss = mean(rowloss)
// ---------------------------------------------------------------------------
__global__ __launch_bounds__(256)
void sae_loss(const float* __restrict__ rowloss, float* __restrict__ loss) {
    __shared__ float red[256];
    float s = 0.0f;
    for (int i = threadIdx.x; i < B_SZ; i += 256) s += rowloss[i];
    red[threadIdx.x] = s;
    __syncthreads();
    for (int st = 128; st > 0; st >>= 1) {
        if (threadIdx.x < st) red[threadIdx.x] += red[threadIdx.x + st];
        __syncthreads();
    }
    if (threadIdx.x == 0) *loss = red[0] / (float)B_SZ;
}

// ---------------------------------------------------------------------------
// Launch: GEMM -> TopK -> Decode -> Loss (stream-ordered, graph-capture safe)
// ---------------------------------------------------------------------------
extern "C" void kernel_launch(void* const* d_in, const int* in_sizes, int n_in,
                              void* d_out, int out_size, void* d_ws, size_t ws_size,
                              hipStream_t stream) {
    const float* x    = (const float*)d_in[0];
    const float* Wenc = (const float*)d_in[1];
    const float* A    = (const float*)d_in[2];
    const float* Bm   = (const float*)d_in[3];
    const float* benc = (const float*)d_in[4];
    const float* bdec = (const float*)d_in[5];
    // d_in[6] is k; fixed to 64 for this problem shape.

    float* out  = (float*)d_out;
    float* loss = out;                                      // [1]
    float* xhat = out + 1;                                  // [4096*5*768]
    float* z    = out + 1 + (size_t)B_SZ * T_POS * D_IN;    // [4096*16384]

    char*  ws      = (char*)d_ws;
    int*   idxL    = (int*)ws;                                        // 1 MB
    float* valL    = (float*)(ws + (size_t)B_SZ * K_SEL * 4);         // 1 MB
    float* rowloss = (float*)(ws + 2 * (size_t)B_SZ * K_SEL * 4);     // 16 KB

    sae_enc_gemm<<<dim3(MDIM / BM, NDIM / BN), 256, 0, stream>>>(x, Wenc, benc, z);
    sae_topk<<<B_SZ, 256, TOPK_LDS_BYTES, stream>>>(z, idxL, valL);
    sae_decode<<<B_SZ, 256, 0, stream>>>(x, A, Bm, bdec, idxL, valL, xhat, rowloss);
    sae_loss<<<1, 256, 0, stream>>>(rowloss, loss);
}
